// NeuralNetwork_65618510348896
// MI455X (gfx1250) — compile-verified, matrix operand order checked by reading, not run on hardware
//
#include <hip/hip_runtime.h>
#include <hip/hip_bf16.h>


typedef _Float16 v16h __attribute__((ext_vector_type(16)));
typedef float    v8f  __attribute__((ext_vector_type(8)));
typedef int      v8i  __attribute__((ext_vector_type(8)));

#define BM 128
#define BN 64


__device__ __forceinline__ void async_b128x2(unsigned lds, const void* g) {
  asm volatile("global_load_async_to_lds_b128 %0, %1, off\n\t"
               "global_load_async_to_lds_b128 %0, %1, off offset:16"
               :: "v"(lds), "v"(g) : "memory");
}
__device__ __forceinline__ void async_b128(unsigned lds, const void* g) {
  asm volatile("global_load_async_to_lds_b128 %0, %1, off"
               :: "v"(lds), "v"(g) : "memory");
}
__device__ __forceinline__ void wait_async(int pending) {
  if (pending)
    asm volatile("s_wait_asynccnt 0x3" ::: "memory");
  else
    asm volatile("s_wait_asynccnt 0x0" ::: "memory");
}
__device__ __forceinline__ unsigned lds_addr(const void* p) {
  return (unsigned)(unsigned long long)p;
}


__device__ __forceinline__ v16h frag_f16(const _Float16* base, int lane) {
  const _Float16* p = base + (size_t)(lane & 15) * 32;
  const int kb = (lane & 16) ? 8 : 0;
  union { v16h v; uint4 q[2]; } u;
  u.q[0] = *reinterpret_cast<const uint4*>(p + kb);
  u.q[1] = *reinterpret_cast<const uint4*>(p + kb + 16);
  return u.v;
}

__device__ __forceinline__ v8i frag_i8(const signed char* base, int lane) {
  const signed char* p = base + (size_t)(lane & 15) * 64;
  const int kb = (lane & 16) ? 8 : 0;
  union { v8i v; uint2 q[4]; } u;
  u.q[0] = *reinterpret_cast<const uint2*>(p + kb);
  u.q[1] = *reinterpret_cast<const uint2*>(p + kb + 16);
  u.q[2] = *reinterpret_cast<const uint2*>(p + kb + 32);
  u.q[3] = *reinterpret_cast<const uint2*>(p + kb + 48);
  return u.v;
}


__global__ void x_to_f16(const float* __restrict__ X, _Float16* __restrict__ O,
                         int K, int Kpad) {
  size_t idx = (size_t)blockIdx.x * blockDim.x + threadIdx.x;
  int c = (int)(idx % (size_t)Kpad);
  size_t r = idx / (size_t)Kpad;
  O[idx] = (c < K) ? (_Float16)X[r * (size_t)K + c] : (_Float16)0.f;
}

__global__ void sign_w_f16(const float* __restrict__ W, _Float16* __restrict__ O,
                           int N, int K, int Kpad) {
  size_t idx = (size_t)blockIdx.x * blockDim.x + threadIdx.x;
  int c = (int)(idx % (size_t)Kpad);
  size_t r = idx / (size_t)Kpad;
  float s = 0.f;
  if ((int)r < N && c < K) {
    float w = W[r * (size_t)K + c];
    s = (w > 0.f) ? 1.f : ((w < 0.f) ? -1.f : 0.f);
  }
  O[idx] = (_Float16)s;
}

__global__ void sign_w_i8(const float* __restrict__ W, signed char* __restrict__ O,
                          int K, int Kpad) {
  size_t idx = (size_t)blockIdx.x * blockDim.x + threadIdx.x;
  int c = (int)(idx % (size_t)Kpad);
  size_t r = idx / (size_t)Kpad;
  signed char s = 0;
  if (c < K) {
    float w = W[r * (size_t)K + c];
    s = (w > 0.f) ? 1 : ((w < 0.f) ? -1 : 0);
  }
  O[idx] = s;
}

__global__ __launch_bounds__(256) void gemm_f16(
    const _Float16* __restrict__ Xh, const _Float16* __restrict__ Wf,
    const float* __restrict__ bias, float* __restrict__ H,
    int Kpad, int N, int Nstride) {
  __shared__ __align__(16) _Float16 As[2][BM][32];
  __shared__ __align__(16) _Float16 Bs[2][BN][32];
  const int tid  = threadIdx.x;
  const int lane = tid & 31;
  const int wave = tid >> 5;
  const int wm = (wave & 3) * 32;
  const int wn = (wave >> 2) * 32;
  const int m0 = blockIdx.x * BM;
  const int n0 = blockIdx.y * BN;

  const int ar = tid >> 1, ac = (tid & 1) * 16;
  const int br = tid >> 2, bc = (tid & 3) * 8;
  const _Float16* aSrc = &Xh[(size_t)(m0 + ar) * Kpad + ac];
  const _Float16* bSrc = &Wf[(size_t)(n0 + br) * Kpad + bc];
  const unsigned aL[2] = { lds_addr(&As[0][ar][ac]), lds_addr(&As[1][ar][ac]) };
  const unsigned bL[2] = { lds_addr(&Bs[0][br][bc]), lds_addr(&Bs[1][br][bc]) };

  v8f acc[2][2] = {};

  const int nt = Kpad / 32;
  async_b128x2(aL[0], aSrc);
  async_b128(bL[0], bSrc);
  for (int t = 0; t < nt; ++t) {
    const int cur = t & 1;
    const bool more = (t + 1 < nt);
    if (more) {
      async_b128x2(aL[cur ^ 1], aSrc + (size_t)(t + 1) * 32);
      async_b128(bL[cur ^ 1], bSrc + (size_t)(t + 1) * 32);
    }
    wait_async(more ? 1 : 0);
    __syncthreads();

    v16h af[2], bf[2];
#pragma unroll
    for (int i = 0; i < 2; ++i) af[i] = frag_f16(&As[cur][wm + i * 16][0], lane);
#pragma unroll
    for (int j = 0; j < 2; ++j) bf[j] = frag_f16(&Bs[cur][wn + j * 16][0], lane);
#pragma unroll
    for (int i = 0; i < 2; ++i)
#pragma unroll
      for (int j = 0; j < 2; ++j)
        acc[i][j] = __builtin_amdgcn_wmma_f32_16x16x32_f16(
            false, af[i], false, bf[j], (short)0, acc[i][j], false, false);
    __syncthreads();
  }

#pragma unroll
  for (int i = 0; i < 2; ++i) {
#pragma unroll
    for (int j = 0; j < 2; ++j) {
      int n = n0 + wn + j * 16 + (lane & 15);
      float bv = (n < N) ? bias[n] : 0.f;
#pragma unroll
      for (int v = 0; v < 8; ++v) {
        int m = m0 + wm + i * 16 + v + ((lane & 16) ? 8 : 0);
        H[(size_t)m * Nstride + n] = acc[i][j][v] + bv;
      }
    }
  }
}

__global__ __launch_bounds__(256) void gemm_i8(
    const signed char* __restrict__ A, const signed char* __restrict__ Bw,
    const float* __restrict__ bias, float* __restrict__ H,
    int Kp, int Nstride) {
  __shared__ __align__(16) signed char As[2][BM][64];
  __shared__ __align__(16) signed char Bs[2][BN][64];
  const int tid  = threadIdx.x;
  const int lane = tid & 31;
  const int wave = tid >> 5;
  const int wm = (wave & 3) * 32;
  const int wn = (wave >> 2) * 32;
  const int m0 = blockIdx.x * BM;
  const int n0 = blockIdx.y * BN;

  const int ar = tid >> 1, ac = (tid & 1) * 32;
  const int br = tid >> 2, bc = (tid & 3) * 16;
  const signed char* aSrc = &A[(size_t)(m0 + ar) * Kp + ac];
  const signed char* bSrc = &Bw[(size_t)(n0 + br) * Kp + bc];
  const unsigned aL[2] = { lds_addr(&As[0][ar][ac]), lds_addr(&As[1][ar][ac]) };
  const unsigned bL[2] = { lds_addr(&Bs[0][br][bc]), lds_addr(&Bs[1][br][bc]) };

  v8i acc[2][2] = {};

  const int nt = Kp / 64;
  async_b128x2(aL[0], aSrc);
  async_b128(bL[0], bSrc);
  for (int t = 0; t < nt; ++t) {
    const int cur = t & 1;
    const bool more = (t + 1 < nt);
    if (more) {
      async_b128x2(aL[cur ^ 1], aSrc + (size_t)(t + 1) * 64);
      async_b128(bL[cur ^ 1], bSrc + (size_t)(t + 1) * 64);
    }
    wait_async(more ? 1 : 0);
    __syncthreads();

    v8i af[2], bf[2];
#pragma unroll
    for (int i = 0; i < 2; ++i) af[i] = frag_i8(&As[cur][wm + i * 16][0], lane);
#pragma unroll
    for (int j = 0; j < 2; ++j) bf[j] = frag_i8(&Bs[cur][wn + j * 16][0], lane);
#pragma unroll
    for (int i = 0; i < 2; ++i)
#pragma unroll
      for (int j = 0; j < 2; ++j)
        acc[i][j] = __builtin_amdgcn_wmma_i32_16x16x64_iu8(
            true, af[i], true, bf[j], acc[i][j], false, false);
    __syncthreads();
  }

#pragma unroll
  for (int i = 0; i < 2; ++i) {
#pragma unroll
    for (int j = 0; j < 2; ++j) {
      int n = n0 + wn + j * 16 + (lane & 15);
      float bv = bias[n];
#pragma unroll
      for (int v = 0; v < 8; ++v) {
        int m = m0 + wm + i * 16 + v + ((lane & 16) ? 8 : 0);
        H[(size_t)m * Nstride + n] = (float)acc[i][j][v] + bv;
      }
    }
  }
}


__global__ void bn_partial(const float* __restrict__ H, float* __restrict__ psum,
                           float* __restrict__ psq, int Nstride, int Nfeat) {
  int f = blockIdx.x * blockDim.x + threadIdx.x;
  if (f >= Nfeat) return;
  size_t r0 = (size_t)blockIdx.y * 256;
  float s = 0.f, ss = 0.f;
  for (int r = 0; r < 256; ++r) {
    float v = H[(r0 + r) * (size_t)Nstride + f];
    s += v;
    ss += v * v;
  }
  psum[(size_t)blockIdx.y * Nfeat + f] = s;
  psq [(size_t)blockIdx.y * Nfeat + f] = ss;
}

__global__ void bn_finalize(const float* __restrict__ psum, const float* __restrict__ psq,
                            const float* __restrict__ gamma, const float* __restrict__ beta,
                            float* __restrict__ scale, float* __restrict__ shift,
                            int Nfeat, int chunks, float invM) {
  int f = blockIdx.x * blockDim.x + threadIdx.x;
  if (f >= Nfeat) return;
  float s = 0.f, ss = 0.f;
  for (int c = 0; c < chunks; ++c) {
    s  += psum[(size_t)c * Nfeat + f];
    ss += psq [(size_t)c * Nfeat + f];
  }
  float mean = s * invM;
  float var = fmaxf(ss * invM - mean * mean, 0.f);
  float sc = gamma[f] * rsqrtf(var + 1e-5f);
  scale[f] = sc;
  shift[f] = beta[f] - mean * sc;
}

__global__ void bn_sign_i8(const float* __restrict__ H, const float* __restrict__ scale,
                           const float* __restrict__ shift, signed char* __restrict__ A,
                           int Npad, int Nfeat) {
  size_t idx = (size_t)blockIdx.x * blockDim.x + threadIdx.x;
  int f = (int)(idx % (size_t)Npad);
  size_t row = idx / (size_t)Npad;
  signed char v = 0;
  if (f < Nfeat) {
    float y = H[row * (size_t)Npad + f] * scale[f] + shift[f];
    v = (y > 0.f) ? 1 : ((y < 0.f) ? -1 : 0);
  }
  A[idx] = v;
}


__global__ __launch_bounds__(256) void fc4_logsoftmax(
    const float* __restrict__ H3, const float* __restrict__ scale,
    const float* __restrict__ shift, const float* __restrict__ W4,
    const float* __restrict__ b4, float* __restrict__ out, int K) {
  __shared__ float Ws[10 * 1024];
  __shared__ float bs[10];
  for (int i = threadIdx.x; i < 10 * 1024; i += 256) Ws[i] = W4[i];
  if (threadIdx.x < 10) bs[threadIdx.x] = b4[threadIdx.x];
  __syncthreads();

  const int wave = threadIdx.x >> 5, lane = threadIdx.x & 31;
  size_t row = (size_t)blockIdx.x * 8 + wave;
  float acc[10] = {};
  for (int k = lane; k < K; k += 32) {
    float h = H3[row * (size_t)K + k] * scale[k] + shift[k];
    h = fminf(1.f, fmaxf(-1.f, h));
#pragma unroll
    for (int o = 0; o < 10; ++o) acc[o] += h * Ws[o * 1024 + k];
  }
#pragma unroll
  for (int o = 0; o < 10; ++o)
    for (int off = 16; off > 0; off >>= 1) acc[o] += __shfl_xor(acc[o], off, 32);

  if (lane == 0) {
    float logits[10], mx = -3.4e38f;
#pragma unroll
    for (int o = 0; o < 10; ++o) {
      logits[o] = acc[o] + bs[o];
      mx = fmaxf(mx, logits[o]);
    }
    float s = 0.f;
#pragma unroll
    for (int o = 0; o < 10; ++o) s += __expf(logits[o] - mx);
    float lse = __logf(s) + mx;
#pragma unroll
    for (int o = 0; o < 10; ++o) out[row * 10 + o] = logits[o] - lse;
  }
}


extern "C" void kernel_launch(void* const* d_in, const int* in_sizes, int n_in,
                              void* d_out, int out_size, void* d_ws, size_t ws_size,
                              hipStream_t stream) {
  (void)in_sizes; (void)n_in; (void)out_size; (void)ws_size;
  const float* x   = (const float*)d_in[0];
  const float* W1  = (const float*)d_in[1];
  const float* b1  = (const float*)d_in[2];
  const float* g1  = (const float*)d_in[3];
  const float* be1 = (const float*)d_in[4];
  const float* W2  = (const float*)d_in[5];
  const float* b2  = (const float*)d_in[6];
  const float* g2  = (const float*)d_in[7];
  const float* be2 = (const float*)d_in[8];
  const float* W3  = (const float*)d_in[9];
  const float* b3  = (const float*)d_in[10];
  const float* g3  = (const float*)d_in[11];
  const float* be3 = (const float*)d_in[12];
  const float* W4  = (const float*)d_in[13];
  const float* b4  = (const float*)d_in[14];
  float* out = (float*)d_out;

  const int M = 65536;

  char* ws = (char*)d_ws;
  size_t off = 0;
  auto take = [&](size_t b) -> char* {
    char* p = ws + off;
    off = (off + b + 255) & ~(size_t)255;
    return p;
  };
  _Float16*    w1f  = (_Float16*)take((size_t)512 * 800 * 2);
  signed char* w2i  = (signed char*)take((size_t)1024 * 512);
  signed char* w3i  = (signed char*)take((size_t)1024 * 1024);
  float*       sc1  = (float*)take(1024 * 4);
  float*       sh1  = (float*)take(1024 * 4);
  float*       sc2  = (float*)take(1024 * 4);
  float*       sh2  = (float*)take(1024 * 4);
  float*       sc3  = (float*)take(1024 * 4);
  float*       sh3  = (float*)take(1024 * 4);
  float*       psum = (float*)take((size_t)256 * 1024 * 4);
  float*       psq  = (float*)take((size_t)256 * 1024 * 4);
  float*       h1   = (float*)take((size_t)M * 512 * 4);
  signed char* A2   = (signed char*)take((size_t)M * 512);
  float*       h2   = (float*)take((size_t)M * 1024 * 4);
  _Float16*    Xh   = (_Float16*)h2;
  signed char* A3   = (signed char*)h1;
  float*       h3   = h2;

  x_to_f16  <<<((size_t)M * 800) / 256, 256, 0, stream>>>(x, Xh, 784, 800);
  sign_w_f16<<<(512 * 800) / 256, 256, 0, stream>>>(W1, w1f, 500, 784, 800);
  sign_w_i8 <<<(1024 * 512) / 256, 256, 0, stream>>>(W2, w2i, 500, 512);
  sign_w_i8 <<<(1024 * 1024) / 256, 256, 0, stream>>>(W3, w3i, 1024, 1024);

  gemm_f16<<<dim3(M / BM, 512 / BN), 256, 0, stream>>>(Xh, w1f, b1, h1,
                                                       800, 500, 512);
  bn_partial <<<dim3(2, 256), 256, 0, stream>>>(h1, psum, psq, 512, 500);
  bn_finalize<<<2, 256, 0, stream>>>(psum, psq, g1, be1, sc1, sh1, 500, 256,
                                     1.f / M);
  bn_sign_i8 <<<((size_t)M * 512) / 256, 256, 0, stream>>>(h1, sc1, sh1, A2, 512, 500);

  gemm_i8<<<dim3(M / BM, 1024 / BN), 256, 0, stream>>>(A2, w2i, b2, h2, 512, 1024);
  bn_partial <<<dim3(4, 256), 256, 0, stream>>>(h2, psum, psq, 1024, 1024);
  bn_finalize<<<4, 256, 0, stream>>>(psum, psq, g2, be2, sc2, sh2, 1024, 256,
                                     1.f / M);
  bn_sign_i8 <<<((size_t)M * 1024) / 256, 256, 0, stream>>>(h2, sc2, sh2, A3, 1024, 1024);

  gemm_i8<<<dim3(M / BM, 1024 / BN), 256, 0, stream>>>(A3, w3i, b3, h3, 1024, 1024);
  bn_partial <<<dim3(4, 256), 256, 0, stream>>>(h3, psum, psq, 1024, 1024);
  bn_finalize<<<4, 256, 0, stream>>>(psum, psq, g3, be3, sc3, sh3, 1024, 256,
                                     1.f / M);

  fc4_logsoftmax<<<M / 8, 256, 0, stream>>>(h3, sc3, sh3, W4, b4, out, 1024);
}
